// Decoder_29300266893979
// MI455X (gfx1250) — compile-verified, matrix operand order checked by reading, not run on hardware
//
#include <hip/hip_runtime.h>
#include <cstdint>
#include <cstddef>

// ---------------------------------------------------------------------------
// Tacotron2-style decoder for MI455X (gfx1250, wave32, WMMA).
// Strategy: LSTM gate GEMMs are weight-bandwidth bound (M=32).  Weights are
// converted once per call to bf16 and pre-packed into the per-lane WMMA
// B-fragment layout; per-step GEMMs then run v_wmma_f32_16x16x32_bf16 with
// pure contiguous 128-bit loads.  All 200 steps are launched on `stream`
// (graph-captured by the harness).
// ---------------------------------------------------------------------------

typedef __attribute__((ext_vector_type(8)))  float  v8f;
typedef __attribute__((ext_vector_type(16))) __bf16 v16bf;

#define FCN   80
#define RR    2
#define QDIM  1024
#define DDIM  1024
#define PDIM  256
#define ADIM  128
#define ENCD  512
#define NFILT 32
#define KSZ   31
#define PADC  15
#define BB    32
#define TIN   256
#define TOUT  200
#define NGATE 4096

// K-tiles (K/32) for the three packed GEMMs
#define KT_ATTN 56   // K = 256+512+1024 = 1792
#define KT_DEC  80   // K = 1024+512+1024 = 2560
#define KT_PROJ 48   // K = 1024+512 = 1536

__device__ __forceinline__ unsigned short f2bf(float f) {
  union { float f; unsigned u; } v; v.f = f;
  unsigned u = v.u;
  u += 0x7FFFu + ((u >> 16) & 1u);      // round-to-nearest-even
  return (unsigned short)(u >> 16);
}
__device__ __forceinline__ float sigf(float x) { return 1.0f / (1.0f + __expf(-x)); }

// Per-lane fragment addressing for 16-bit WMMA operands (16x32 A / 32x16 B):
//   lane 0-15 -> row/col = lane, K-base 0;  lane 16-31 -> row/col = lane-16, K-base 8
//   element j (0..15): K = base + j + (j & 8)
__device__ __forceinline__ int frag_k(int lane, int j) {
  return ((lane & 16) >> 1) + j + (j & 8);
}

// ------------------------------- prep kernels ------------------------------

__global__ void k_zero(float* __restrict__ p, int n) {
  int i = blockIdx.x * blockDim.x + threadIdx.x;
  if (i < n) p[i] = 0.0f;
}

// Pack [rows0 | rows1] x N fp32 weights into bf16 WMMA B-fragments.
__global__ void k_pack_B(unsigned short* __restrict__ dst,
                         const float* __restrict__ src0, int rows0,
                         const float* __restrict__ src1,
                         int N, int Ktiles, int total) {
  int e = blockIdx.x * blockDim.x + threadIdx.x;
  if (e >= total) return;
  int j = e & 15, lane = (e >> 4) & 31, frag = e >> 9;
  int kt = frag % Ktiles, ntile = frag / Ktiles;
  int n = ntile * 16 + (lane & 15);
  int k = kt * 32 + frag_k(lane, j);
  float v = (k < rows0) ? src0[k * N + n] : src1[(k - rows0) * N + n];
  dst[e] = f2bf(v);
}

// Pack concatenated activations [s0|s1|s2] (each [32][len]) into A-fragments.
__global__ void k_pack_A(unsigned short* __restrict__ dst,
                         const float* __restrict__ s0, int l0,
                         const float* __restrict__ s1, int l1,
                         const float* __restrict__ s2, int l2,
                         int Ktiles, int total) {
  int e = blockIdx.x * blockDim.x + threadIdx.x;
  if (e >= total) return;
  int j = e & 15, lane = (e >> 4) & 31, frag = e >> 9;
  int kt = frag % Ktiles, mtile = frag / Ktiles;
  int m = mtile * 16 + (lane & 15);
  int k = kt * 32 + frag_k(lane, j);
  float v;
  if (k < l0)            v = s0[m * l0 + k];
  else if (k < l0 + l1)  v = s1[m * l1 + (k - l0)];
  else                   v = s2[m * l2 + (k - l0 - l1)];
  dst[e] = f2bf(v);
}

// prenet for all 200 steps: pre[t][b][:] = relu(relu(x @ W1) @ W2)
__global__ void k_prenet(const float* __restrict__ memories,
                         const float* __restrict__ W1,
                         const float* __restrict__ W2,
                         float* __restrict__ pre) {
  int t = blockIdx.x / BB, b = blockIdx.x % BB, tid = threadIdx.x;
  __shared__ float sx[FCN];
  __shared__ float sh[PDIM];
  if (tid < FCN)
    sx[tid] = (t == 0) ? 0.0f : memories[(b * (TOUT * RR) + (2 * t - 1)) * FCN + tid];
  __syncthreads();
  float a = 0.0f;
  for (int k = 0; k < FCN; ++k) a += sx[k] * W1[k * PDIM + tid];
  sh[tid] = fmaxf(a, 0.0f);
  __syncthreads();
  float o = 0.0f;
  for (int k = 0; k < PDIM; ++k) o += sh[k] * W2[k * PDIM + tid];
  pre[((size_t)t * BB + b) * PDIM + tid] = fmaxf(o, 0.0f);
}

// processed_inputs[b][t][:] = inputs[b][t][:] @ W_inputs
__global__ void k_procin(const float* __restrict__ inputs,
                         const float* __restrict__ Wi,
                         float* __restrict__ pi) {
  int bt = blockIdx.x, tid = threadIdx.x;   // 128 threads
  __shared__ float sx[ENCD];
  for (int i = tid; i < ENCD; i += 128) sx[i] = inputs[(size_t)bt * ENCD + i];
  __syncthreads();
  float s = 0.0f;
  for (int d = 0; d < ENCD; ++d) s += sx[d] * Wi[d * ADIM + tid];
  pi[(size_t)bt * ADIM + tid] = s;
}

// ------------------------------- WMMA GEMM ---------------------------------

// out[32][N] = A(32 x 32*Ktiles, packed bf16 frags) x B(packed bf16 frags)
__global__ void k_gemm(const unsigned short* __restrict__ Af,
                       const unsigned short* __restrict__ Bf,
                       float* __restrict__ out,
                       int Ktiles, int Ntiles, int N) {
  int wave = threadIdx.x >> 5, lane = threadIdx.x & 31;
  int mtile = wave & 1;
  int ntile = blockIdx.x * 4 + (wave >> 1);
  if (ntile >= Ntiles) return;
  const v16bf* Ap = (const v16bf*)Af + (size_t)mtile * Ktiles * 32 + lane;
  const v16bf* Bp = (const v16bf*)Bf + (size_t)ntile * Ktiles * 32 + lane;
  v8f acc = {};
  for (int kt = 0; kt < Ktiles; ++kt) {
    __builtin_prefetch((const char*)(Bp + (kt + 8) * 32), 0, 1);
    v16bf a = Ap[kt * 32];
    v16bf b = Bp[kt * 32];
    acc = __builtin_amdgcn_wmma_f32_16x16x32_bf16(
        false, a, false, b, (short)0, acc, false, false);
  }
  int m = mtile * 16 + ((lane >> 4) << 3);
  int n = ntile * 16 + (lane & 15);
#pragma unroll
  for (int r = 0; r < 8; ++r) out[(size_t)(m + r) * N + n] = acc[r];
}

// projection GEMM (N=160) fused with bias + scatter into d_out layout
__global__ void k_gemm_proj(const unsigned short* __restrict__ Af,
                            const unsigned short* __restrict__ Bf,
                            const float* __restrict__ proj_b,
                            float* __restrict__ dec_out_buf,
                            float* __restrict__ outp,   // outputs [B][80][400]
                            int t) {
  int wave = threadIdx.x >> 5, lane = threadIdx.x & 31;
  int mtile = wave & 1;
  int ntile = blockIdx.x * 4 + (wave >> 1);
  if (ntile >= 10) return;
  const v16bf* Ap = (const v16bf*)Af + (size_t)mtile * KT_PROJ * 32 + lane;
  const v16bf* Bp = (const v16bf*)Bf + (size_t)ntile * KT_PROJ * 32 + lane;
  v8f acc = {};
  for (int kt = 0; kt < KT_PROJ; ++kt) {
    v16bf a = Ap[kt * 32];
    v16bf b = Bp[kt * 32];
    acc = __builtin_amdgcn_wmma_f32_16x16x32_bf16(
        false, a, false, b, (short)0, acc, false, false);
  }
  int m = mtile * 16 + ((lane >> 4) << 3);
  int n = ntile * 16 + (lane & 15);
  int f = n % FCN, tt = 2 * t + n / FCN;
#pragma unroll
  for (int r = 0; r < 8; ++r) {
    int b = m + r;
    float val = acc[r] + proj_b[n];
    dec_out_buf[b * (FCN * RR) + n] = val;
    outp[(size_t)b * FCN * (TOUT * RR) + (size_t)f * (TOUT * RR) + tt] = val;
  }
}

// ------------------------------- LSTM / misc -------------------------------

__global__ void k_lstm(const float* __restrict__ gates,
                       const float* __restrict__ bias,
                       float* __restrict__ h, float* __restrict__ c) {
  int idx = blockIdx.x * blockDim.x + threadIdx.x;
  if (idx >= BB * QDIM) return;
  int b = idx >> 10, n = idx & 1023;
  const float* gb = gates + (size_t)b * NGATE;
  float ig = gb[n]            + bias[n];
  float fg = gb[1024 + n]     + bias[1024 + n];
  float gg = gb[2048 + n]     + bias[2048 + n];
  float og = gb[3072 + n]     + bias[3072 + n];
  float cc = sigf(fg) * c[idx] + sigf(ig) * tanhf(gg);
  c[idx] = cc;
  h[idx] = sigf(og) * tanhf(cc);
}

// attention: pq, location conv, energies, softmax, awc update, context.
// one block per batch element, 256 threads (= T_in)
__global__ void k_attention(const float* __restrict__ qh,
                            const float* __restrict__ Wq,
                            const float* __restrict__ conv_w,
                            const float* __restrict__ W_loc,
                            const float* __restrict__ v_w,
                            const float* __restrict__ v_b,
                            const unsigned char* __restrict__ mask,
                            const float* __restrict__ pi,
                            const float* __restrict__ inputs,
                            float* __restrict__ aw,
                            float* __restrict__ awc,
                            float* __restrict__ ctx,
                            float* __restrict__ out_align,
                            int t) {
  __shared__ float s_pq[ADIM];
  __shared__ float s_aw[TIN], s_awc[TIN], s_al[TIN], s_red[TIN];
  __shared__ float s_conv[NFILT * 2 * KSZ];   // 1984
  __shared__ float s_wloc[NFILT * ADIM];      // 4096
  __shared__ float s_vw[ADIM];
  int b = blockIdx.x, tid = threadIdx.x;
  s_aw[tid]  = aw[b * TIN + tid];
  s_awc[tid] = awc[b * TIN + tid];
  for (int i = tid; i < NFILT * 2 * KSZ; i += 256) s_conv[i] = conv_w[i];
  for (int i = tid; i < NFILT * ADIM; i += 256)    s_wloc[i] = W_loc[i];
  if (tid < ADIM) {
    s_vw[tid] = v_w[tid];
    float s = 0.0f;
    for (int k = 0; k < QDIM; ++k) s += qh[b * QDIM + k] * Wq[k * ADIM + tid];
    s_pq[tid] = s;
  }
  __syncthreads();

  // location conv at position tid (correlation, pad 15)
  float loc[NFILT];
#pragma unroll
  for (int f = 0; f < NFILT; ++f) loc[f] = 0.0f;
  for (int c = 0; c < 2; ++c) {
    const float* src = c ? s_awc : s_aw;
    for (int k = 0; k < KSZ; ++k) {
      int p = tid + k - PADC;
      float x = (p >= 0 && p < TIN) ? src[p] : 0.0f;
      const float* wrow = &s_conv[c * KSZ + k];
#pragma unroll
      for (int f = 0; f < NFILT; ++f) loc[f] += x * wrow[f * (2 * KSZ)];
    }
  }
  // energies
  float e = 0.0f;
  for (int a = 0; a < ADIM; ++a) {
    float pa = 0.0f;
#pragma unroll
    for (int f = 0; f < NFILT; ++f) pa += loc[f] * s_wloc[f * ADIM + a];
    float u = tanhf(s_pq[a] + pa + pi[((size_t)b * TIN + tid) * ADIM + a]);
    e += u * s_vw[a];
  }
  e += v_b[0];
  if (!mask[b * TIN + tid]) e = -1.0e9f;

  // softmax over T_in
  s_red[tid] = e; __syncthreads();
  for (int o = 128; o > 0; o >>= 1) {
    if (tid < o) s_red[tid] = fmaxf(s_red[tid], s_red[tid + o]);
    __syncthreads();
  }
  float mx = s_red[0]; __syncthreads();
  float ex = __expf(e - mx);
  s_red[tid] = ex; __syncthreads();
  for (int o = 128; o > 0; o >>= 1) {
    if (tid < o) s_red[tid] += s_red[tid + o];
    __syncthreads();
  }
  float al = ex / s_red[0];
  s_al[tid] = al;
  aw[b * TIN + tid]  = al;
  awc[b * TIN + tid] = s_awc[tid] + al;
  out_align[((size_t)b * TOUT + t) * TIN + tid] = al;
  __syncthreads();

  // context = align . inputs
  for (int d = tid; d < ENCD; d += 256) {
    float s = 0.0f;
    for (int tt = 0; tt < TIN; ++tt)
      s += s_al[tt] * inputs[((size_t)b * TIN + tt) * ENCD + d];
    ctx[b * ENCD + d] = s;
  }
}

__global__ void k_stop(const float* __restrict__ dh,
                       const float* __restrict__ dec_out_buf,
                       const float* __restrict__ stop_w,
                       const float* __restrict__ stop_b,
                       float* __restrict__ out_stop, int t) {
  int b = threadIdx.x;
  if (b >= BB) return;
  float s = stop_b[0];
  for (int k = 0; k < DDIM; ++k) s += dh[b * DDIM + k] * stop_w[k];
  for (int j = 0; j < FCN * RR; ++j) s += dec_out_buf[b * (FCN * RR) + j] * stop_w[DDIM + j];
  out_stop[b * TOUT + t] = s;
}

// ------------------------------- host launcher -----------------------------

extern "C" void kernel_launch(void* const* d_in, const int* in_sizes, int n_in,
                              void* d_out, int out_size, void* d_ws, size_t ws_size,
                              hipStream_t stream) {
  (void)in_sizes; (void)n_in; (void)out_size; (void)ws_size;
  const float* inputs    = (const float*)d_in[0];
  const float* memories  = (const float*)d_in[1];
  const unsigned char* mask = (const unsigned char*)d_in[2];
  const float* prenet_w1 = (const float*)d_in[3];
  const float* prenet_w2 = (const float*)d_in[4];
  const float* attn_wih  = (const float*)d_in[5];
  const float* attn_whh  = (const float*)d_in[6];
  const float* attn_b    = (const float*)d_in[7];
  const float* W_query   = (const float*)d_in[8];
  const float* W_inputs  = (const float*)d_in[9];
  const float* conv_w    = (const float*)d_in[10];
  const float* W_loc     = (const float*)d_in[11];
  const float* v_w       = (const float*)d_in[12];
  const float* v_b       = (const float*)d_in[13];
  const float* dec_wih   = (const float*)d_in[14];
  const float* dec_whh   = (const float*)d_in[15];
  const float* dec_b     = (const float*)d_in[16];
  const float* proj_w    = (const float*)d_in[17];
  const float* proj_b    = (const float*)d_in[18];
  const float* stop_w    = (const float*)d_in[19];
  const float* stop_b    = (const float*)d_in[20];

  float* out       = (float*)d_out;
  float* out_align = out + (size_t)BB * FCN * TOUT * RR;    // 1,024,000
  float* out_stop  = out_align + (size_t)BB * TOUT * TIN;   // +1,638,400

  char* base = (char*)d_ws;
  size_t off = 0;
  auto alloc = [&](size_t bytes) -> char* {
    off = (off + 255) & ~(size_t)255;
    char* p = base + off;
    off += bytes;
    return p;
  };

  // contiguous state block (zeroed each call)
  const int stateN = 4 * BB * QDIM + BB * ENCD + 2 * BB * TIN;
  float* states = (float*)alloc((size_t)stateN * 4);
  float* qh = states;
  float* qc = qh + BB * QDIM;
  float* dh = qc + BB * QDIM;
  float* dc = dh + BB * QDIM;
  float* ctx = dc + BB * QDIM;
  float* aw  = ctx + BB * ENCD;
  float* awc = aw + BB * TIN;

  float* gates       = (float*)alloc((size_t)BB * NGATE * 4);
  float* pre         = (float*)alloc((size_t)TOUT * BB * PDIM * 4);
  float* pi          = (float*)alloc((size_t)BB * TIN * ADIM * 4);
  float* dec_out_buf = (float*)alloc((size_t)BB * FCN * RR * 4);

  unsigned short* WgA = (unsigned short*)alloc((size_t)(KT_ATTN * 32) * NGATE * 2);
  unsigned short* WgD = (unsigned short*)alloc((size_t)(KT_DEC * 32) * NGATE * 2);
  unsigned short* WgP = (unsigned short*)alloc((size_t)(KT_PROJ * 32) * 160 * 2);
  unsigned short* AfA = (unsigned short*)alloc((size_t)2 * KT_ATTN * 512 * 2);
  unsigned short* AfD = (unsigned short*)alloc((size_t)2 * KT_DEC * 512 * 2);
  unsigned short* AfP = (unsigned short*)alloc((size_t)2 * KT_PROJ * 512 * 2);

  // ---- one-time prep ----
  k_zero<<<(stateN + 255) / 256, 256, 0, stream>>>(states, stateN);
  {
    int total = KT_ATTN * 32 * NGATE;
    k_pack_B<<<(total + 255) / 256, 256, 0, stream>>>(WgA, attn_wih, PDIM + ENCD,
                                                      attn_whh, NGATE, KT_ATTN, total);
  }
  {
    int total = KT_DEC * 32 * NGATE;
    k_pack_B<<<(total + 255) / 256, 256, 0, stream>>>(WgD, dec_wih, QDIM + ENCD,
                                                      dec_whh, NGATE, KT_DEC, total);
  }
  {
    int total = KT_PROJ * 32 * 160;
    k_pack_B<<<(total + 255) / 256, 256, 0, stream>>>(WgP, proj_w, KT_PROJ * 32,
                                                      proj_w, 160, KT_PROJ, total);
  }
  k_prenet<<<TOUT * BB, 256, 0, stream>>>(memories, prenet_w1, prenet_w2, pre);
  k_procin<<<BB * TIN, 128, 0, stream>>>(inputs, W_inputs, pi);

  // ---- 200 sequential decode steps ----
  for (int t = 0; t < TOUT; ++t) {
    {
      int total = 2 * KT_ATTN * 512;
      k_pack_A<<<(total + 255) / 256, 256, 0, stream>>>(
          AfA, pre + (size_t)t * BB * PDIM, PDIM, ctx, ENCD, qh, QDIM, KT_ATTN, total);
    }
    k_gemm<<<64, 256, 0, stream>>>(AfA, WgA, gates, KT_ATTN, NGATE / 16, NGATE);
    k_lstm<<<(BB * QDIM + 255) / 256, 256, 0, stream>>>(gates, attn_b, qh, qc);

    k_attention<<<BB, 256, 0, stream>>>(qh, W_query, conv_w, W_loc, v_w, v_b,
                                        mask, pi, inputs, aw, awc, ctx, out_align, t);

    {
      int total = 2 * KT_DEC * 512;
      k_pack_A<<<(total + 255) / 256, 256, 0, stream>>>(
          AfD, qh, QDIM, ctx, ENCD, dh, DDIM, KT_DEC, total);
    }
    k_gemm<<<64, 256, 0, stream>>>(AfD, WgD, gates, KT_DEC, NGATE / 16, NGATE);
    k_lstm<<<(BB * DDIM + 255) / 256, 256, 0, stream>>>(gates, dec_b, dh, dc);

    {
      int total = 2 * KT_PROJ * 512;
      k_pack_A<<<(total + 255) / 256, 256, 0, stream>>>(
          AfP, dh, DDIM, ctx, ENCD, ctx, 1, KT_PROJ, total);
    }
    k_gemm_proj<<<3, 256, 0, stream>>>(AfP, WgP, proj_b, dec_out_buf, out, t);
    k_stop<<<1, 32, 0, stream>>>(dh, dec_out_buf, stop_w, stop_b, out_stop, t);
  }
}
